// SparseDimensionsLinear_12979391169361
// MI455X (gfx1250) — matrix operand order repaired and checked by reading, not compiled, not run on hardware
//
#include <hip/hip_runtime.h>

// ---- CDNA5 (gfx1250) types -------------------------------------------------
typedef __attribute__((ext_vector_type(16))) __bf16    v16bf;
typedef __attribute__((ext_vector_type(8)))  float     v8f;
typedef __attribute__((ext_vector_type(4)))  float     v4f;
typedef __attribute__((ext_vector_type(4)))  unsigned  v4u;
typedef __attribute__((ext_vector_type(2)))  unsigned  v2u;
typedef __attribute__((ext_vector_type(4)))  int       v4i;

// pointer flavors for the async-to-LDS builtin:
// param0 = v4i in AS1 (__device__), param1 = v4i in AS3 (__shared__)
typedef __attribute__((address_space(1))) v4i v4i_g;
typedef __attribute__((address_space(3))) v4i v4i_l;

__device__ __forceinline__ v4i_g* to_global_v4i(const void* p) {
  return (v4i_g*)(unsigned long long)p;           // AS1 pointers are 64-bit
}
__device__ __forceinline__ v4i_l* to_lds_v4i(void* p) {
  return (v4i_l*)(unsigned)(unsigned long long)p; // AS3 pointers are 32-bit
}

// ---- problem constants -----------------------------------------------------
#define M_TOTAL 16384   // B * S = 4 * 4096
#define K_DIM   2048    // KEPT_IN
#define N_DIM   2048    // KEPT_OUT
#define IN_F    4096
#define OUT_F   4096

// ---- tiling ----------------------------------------------------------------
#define BM   128
#define BN   128
#define BK   32
#define LDSS 40                 // bf16 elems per LDS row (80B => 20-bank stride, conflict-free b128)
#define KTILES (K_DIM / BK)     // 64
#define TILE_ELEMS (BM * LDSS)  // elements per LDS buffer

// gfx1250 async global->LDS copy (ASYNCcnt path), compile-safe guard
#define HAS_ASYNC_LDS __has_builtin(__builtin_amdgcn_global_load_async_to_lds_b128)

union FragU { v4u u[2]; v16bf v; };

__device__ __forceinline__ unsigned f32_bf16(float f) {
  unsigned u = __float_as_uint(f);
  return (u + 0x7FFFu + ((u >> 16) & 1u)) >> 16;   // round-to-nearest-even
}
__device__ __forceinline__ unsigned pack2(float lo, float hi) {
  return f32_bf16(lo) | (f32_bf16(hi) << 16);
}

// ---- kernel: nontemporal zero fill of the full output ----------------------
__global__ void zero_fill_kernel(float* __restrict__ out, int n4) {
  int i = blockIdx.x * blockDim.x + threadIdx.x;
  if (i < n4) {
    v4f z = {0.f, 0.f, 0.f, 0.f};
    __builtin_nontemporal_store(z, ((v4f*)out) + i);
  }
}

// ---- pass 1a: gather input columns + convert fp32 -> bf16 ------------------
__global__ __launch_bounds__(256) void convert_x_kernel(
    const float* __restrict__ batch, const int* __restrict__ in_idx,
    unsigned short* __restrict__ Xb)
{
  int id  = blockIdx.x * blockDim.x + threadIdx.x;   // [0, M*K/8)
  int row = id >> 8;                                 // K/8 = 256 segs per row
  int seg = id & 255;
  const float* bp = batch + (size_t)row * IN_F;
  v4i i0 = *(const v4i*)&in_idx[seg * 8];
  v4i i1 = *(const v4i*)&in_idx[seg * 8 + 4];
  v4u p;
  p.x = pack2(bp[i0.x], bp[i0.y]);
  p.y = pack2(bp[i0.z], bp[i0.w]);
  p.z = pack2(bp[i1.x], bp[i1.y]);
  p.w = pack2(bp[i1.z], bp[i1.w]);
  __builtin_nontemporal_store(p, (v4u*)&Xb[(size_t)row * K_DIM + seg * 8]);
}

// ---- pass 1b: convert weight fp32 -> bf16 ----------------------------------
__global__ __launch_bounds__(256) void convert_w_kernel(
    const float* __restrict__ W, unsigned short* __restrict__ Wb)
{
  int id  = blockIdx.x * blockDim.x + threadIdx.x;   // [0, N*K/8)
  int row = id >> 8;
  int seg = id & 255;
  const float* wp = W + (size_t)row * K_DIM + seg * 8;
  v4f a = *(const v4f*)wp;
  v4f b = *(const v4f*)(wp + 4);
  v4u p;
  p.x = pack2(a.x, a.y);  p.y = pack2(a.z, a.w);
  p.z = pack2(b.x, b.y);  p.w = pack2(b.z, b.w);
  *(v4u*)&Wb[(size_t)row * K_DIM + seg * 8] = p;
}

__device__ __forceinline__ void wait_stage() {
#if HAS_ASYNC_LDS
 #if __has_builtin(__builtin_amdgcn_s_wait_asynccnt)
  __builtin_amdgcn_s_wait_asynccnt(0);
 #else
  asm volatile("s_wait_asynccnt 0" ::: "memory");
 #endif
#endif
  __syncthreads();
}

// ---- shared WMMA micro-kernel (reads one staged buffer) --------------------
__device__ __forceinline__ void compute_frags(
    const unsigned short* __restrict__ aB,   // per-lane A fragment base
    const unsigned short* __restrict__ bB,   // per-lane B fragment base
    v8f (&acc)[4][2])
{
  // CDNA5 16-bit A fragment: lanes 0-15 hold K[c..c+7] (v0..3), K[c+16..c+23]
  // (v4..7) with c=0; lanes 16-31 use c=8.  B: same K order as A.
  FragU af[4], bfr[2];
  #pragma unroll
  for (int mi = 0; mi < 4; ++mi) {
    const unsigned short* p = aB + mi * 16 * LDSS;
    af[mi].u[0] = *(const v4u*)p;
    af[mi].u[1] = *(const v4u*)(p + 16);
  }
  #pragma unroll
  for (int ni = 0; ni < 2; ++ni) {
    const unsigned short* p = bB + ni * 16 * LDSS;
    bfr[ni].u[0] = *(const v4u*)p;        // K[c..c+7], K[c+16..c+23] like A
    bfr[ni].u[1] = *(const v4u*)(p + 16);
  }
  #pragma unroll
  for (int mi = 0; mi < 4; ++mi)
    #pragma unroll
    for (int ni = 0; ni < 2; ++ni)
      acc[mi][ni] = __builtin_amdgcn_wmma_f32_16x16x32_bf16(
          false, af[mi].v, false, bfr[ni].v, (short)0, acc[mi][ni], false, false);
}

// ---- pass 2: pure bf16 WMMA GEMM, async double-buffered LDS ----------------
__global__ __launch_bounds__(256) void gemm_bf16_wmma_kernel(
    const unsigned short* __restrict__ Xb,   // [M, K] bf16
    const unsigned short* __restrict__ Wb,   // [N, K] bf16
    const float* __restrict__ bias,
    const int*   __restrict__ out_idx,
    float*       __restrict__ out)
{
  __shared__ __align__(16) unsigned short sA[2][TILE_ELEMS];  // 2 x 10 KB
  __shared__ __align__(16) unsigned short sB[2][TILE_ELEMS];  // 2 x 10 KB

  const int tid    = threadIdx.x;
  const int wave   = tid >> 5;
  const int lane   = tid & 31;
  const int lane16 = lane & 15;
  const int laneHi = lane >> 4;

  const int nBlk = blockIdx.x * BN;
  const int mBlk = blockIdx.y * BM;
  const int waveM = wave >> 2;   // 0..1 -> 64 M rows
  const int waveN = wave & 3;    // 0..3 -> 32 N cols

  // staging: each lane moves one 16B chunk, 2 chunks per matrix per wave
  const int id0  = wave * 64 + lane;        // wave*64 + 0..31 (max 479); id1 fills the gaps to 511
  const int id1  = id0 + 32;
  const int row0 = id0 >> 2, ch0 = id0 & 3;
  const int row1 = id1 >> 2, ch1 = id1 & 3;
  const int gOff0 = row0 * K_DIM + ch0 * 8; // 32-bit lane offset (elements)
  const int gOff1 = row1 * K_DIM + ch1 * 8;
  const int lOff0 = row0 * LDSS + ch0 * 8;
  const int lOff1 = row1 * LDSS + ch1 * 8;

  // scalar-advancing global bases (SADDR form friendly)
  const unsigned short* gAk = Xb + (size_t)mBlk * K_DIM;
  const unsigned short* gBk = Wb + (size_t)nBlk * K_DIM;

  // per-lane fragment offsets within one buffer (elements)
  const int aOff = (waveM * 64 + lane16) * LDSS + laneHi * 8;
  const int bOff = (waveN * 32 + lane16) * LDSS + laneHi * 8;

  v8f zero8 = {0.f,0.f,0.f,0.f,0.f,0.f,0.f,0.f};
  v8f acc[4][2];
  #pragma unroll
  for (int mi = 0; mi < 4; ++mi)
    #pragma unroll
    for (int ni = 0; ni < 2; ++ni)
      acc[mi][ni] = zero8;

  // stage k-tile 0 into buffer 0
#if HAS_ASYNC_LDS
  __builtin_amdgcn_global_load_async_to_lds_b128(to_global_v4i(gAk + gOff0), to_lds_v4i(&sA[0][lOff0]), 0, 0);
  __builtin_amdgcn_global_load_async_to_lds_b128(to_global_v4i(gAk + gOff1), to_lds_v4i(&sA[0][lOff1]), 0, 0);
  __builtin_amdgcn_global_load_async_to_lds_b128(to_global_v4i(gBk + gOff0), to_lds_v4i(&sB[0][lOff0]), 0, 0);
  __builtin_amdgcn_global_load_async_to_lds_b128(to_global_v4i(gBk + gOff1), to_lds_v4i(&sB[0][lOff1]), 0, 0);
#else
  *(v4u*)&sA[0][lOff0] = *(const v4u*)(gAk + gOff0);
  *(v4u*)&sA[0][lOff1] = *(const v4u*)(gAk + gOff1);
  *(v4u*)&sB[0][lOff0] = *(const v4u*)(gBk + gOff0);
  *(v4u*)&sB[0][lOff1] = *(const v4u*)(gBk + gOff1);
#endif
  gAk += BK; gBk += BK;
  wait_stage();

  #pragma unroll 1
  for (int kt = 0; kt < KTILES; ++kt) {
    const int cur = kt & 1;
    const int nxt = cur ^ 1;
    if (kt + 1 < KTILES) {
#if HAS_ASYNC_LDS
      __builtin_amdgcn_global_load_async_to_lds_b128(to_global_v4i(gAk + gOff0), to_lds_v4i(&sA[nxt][lOff0]), 0, 0);
      __builtin_amdgcn_global_load_async_to_lds_b128(to_global_v4i(gAk + gOff1), to_lds_v4i(&sA[nxt][lOff1]), 0, 0);
      __builtin_amdgcn_global_load_async_to_lds_b128(to_global_v4i(gBk + gOff0), to_lds_v4i(&sB[nxt][lOff0]), 0, 0);
      __builtin_amdgcn_global_load_async_to_lds_b128(to_global_v4i(gBk + gOff1), to_lds_v4i(&sB[nxt][lOff1]), 0, 0);
#else
      *(v4u*)&sA[nxt][lOff0] = *(const v4u*)(gAk + gOff0);
      *(v4u*)&sA[nxt][lOff1] = *(const v4u*)(gAk + gOff1);
      *(v4u*)&sB[nxt][lOff0] = *(const v4u*)(gBk + gOff0);
      *(v4u*)&sB[nxt][lOff1] = *(const v4u*)(gBk + gOff1);
#endif
      gAk += BK; gBk += BK;
    }
    compute_frags(&sA[cur][aOff], &sB[cur][bOff], acc);
    wait_stage();
  }

  // epilogue: C/D layout -> N = lane%16, M = vgpr + 8*(lane>=16)
  #pragma unroll
  for (int ni = 0; ni < 2; ++ni) {
    const int nG     = nBlk + waveN * 32 + ni * 16 + lane16;
    const int colOut = out_idx[nG];
    const float bv   = bias[nG];
    #pragma unroll
    for (int mi = 0; mi < 4; ++mi) {
      const int mG0 = mBlk + waveM * 64 + mi * 16 + laneHi * 8;
      float* op = out + (size_t)mG0 * OUT_F + colOut;
      #pragma unroll
      for (int i = 0; i < 8; ++i) {
        float r = acc[mi][ni][i] + bv;
        __builtin_nontemporal_store(r, op + (size_t)i * OUT_F);
      }
    }
  }
}

// ======================= fused fallback (small workspace) ===================
__device__ __forceinline__ void load_tiles_fused(
    const float* __restrict__ batch, const float* __restrict__ weight,
    const int* __restrict__ in_idx,
    int mBlk, int nBlk, int k0, int kg, int rb,
    float (&aReg)[4][4], float (&bReg)[4][4])
{
  v4i idx = *(const v4i*)&in_idx[k0 + kg * 4];
  #pragma unroll
  for (int r = 0; r < 4; ++r) {
    const float* bp = batch + (size_t)(mBlk + rb + 32 * r) * IN_F;
    aReg[r][0] = bp[idx.x]; aReg[r][1] = bp[idx.y];
    aReg[r][2] = bp[idx.z]; aReg[r][3] = bp[idx.w];
  }
  #pragma unroll
  for (int r = 0; r < 4; ++r) {
    v4f w4 = *(const v4f*)&weight[(size_t)(nBlk + rb + 32 * r) * K_DIM + k0 + kg * 4];
    bReg[r][0] = w4.x; bReg[r][1] = w4.y; bReg[r][2] = w4.z; bReg[r][3] = w4.w;
  }
}

__device__ __forceinline__ void store_tiles_fused(
    unsigned short* __restrict__ sA, unsigned short* __restrict__ sB,
    int kg, int rb, float (&aReg)[4][4], float (&bReg)[4][4])
{
  #pragma unroll
  for (int r = 0; r < 4; ++r) {
    v2u pa = { pack2(aReg[r][0], aReg[r][1]), pack2(aReg[r][2], aReg[r][3]) };
    *(v2u*)&sA[(rb + 32 * r) * LDSS + kg * 4] = pa;
    v2u pb = { pack2(bReg[r][0], bReg[r][1]), pack2(bReg[r][2], bReg[r][3]) };
    *(v2u*)&sB[(rb + 32 * r) * LDSS + kg * 4] = pb;
  }
}

__global__ __launch_bounds__(256) void sparse_linear_wmma_fused_kernel(
    const float* __restrict__ batch, const float* __restrict__ weight,
    const float* __restrict__ bias, const int* __restrict__ in_idx,
    const int* __restrict__ out_idx, float* __restrict__ out)
{
  __shared__ __align__(16) unsigned short sA[TILE_ELEMS];
  __shared__ __align__(16) unsigned short sB[TILE_ELEMS];

  const int tid = threadIdx.x;
  const int wave = tid >> 5, lane = tid & 31;
  const int lane16 = lane & 15, laneHi = lane >> 4;
  const int nBlk = blockIdx.x * BN, mBlk = blockIdx.y * BM;
  const int waveM = wave >> 2, waveN = wave & 3;
  const int kg = tid & 7, rb = tid >> 3;

  const int aOff = (waveM * 64 + lane16) * LDSS + laneHi * 8;
  const int bOff = (waveN * 32 + lane16) * LDSS + laneHi * 8;

  float aReg[4][4], bReg[4][4];
  v8f zero8 = {0.f,0.f,0.f,0.f,0.f,0.f,0.f,0.f};
  v8f acc[4][2];
  #pragma unroll
  for (int mi = 0; mi < 4; ++mi)
    #pragma unroll
    for (int ni = 0; ni < 2; ++ni)
      acc[mi][ni] = zero8;

  load_tiles_fused(batch, weight, in_idx, mBlk, nBlk, 0, kg, rb, aReg, bReg);
  store_tiles_fused(sA, sB, kg, rb, aReg, bReg);
  __syncthreads();

  #pragma unroll 1
  for (int kt = 0; kt < KTILES; ++kt) {
    if (kt + 1 < KTILES)
      load_tiles_fused(batch, weight, in_idx, mBlk, nBlk, (kt + 1) * BK, kg, rb, aReg, bReg);
    compute_frags(&sA[aOff], &sB[bOff], acc);
    __syncthreads();
    if (kt + 1 < KTILES) {
      store_tiles_fused(sA, sB, kg, rb, aReg, bReg);
      __syncthreads();
    }
  }

  #pragma unroll
  for (int ni = 0; ni < 2; ++ni) {
    const int nG     = nBlk + waveN * 32 + ni * 16 + lane16;
    const int colOut = out_idx[nG];
    const float bv   = bias[nG];
    #pragma unroll
    for (int mi = 0; mi < 4; ++mi) {
      const int mG0 = mBlk + waveM * 64 + mi * 16 + laneHi * 8;
      float* op = out + (size_t)mG0 * OUT_F + colOut;
      #pragma unroll
      for (int i = 0; i < 8; ++i) {
        float r = acc[mi][ni][i] + bv;
        __builtin_nontemporal_store(r, op + (size_t)i * OUT_F);
      }
    }
  }
}

// ---- launch ----------------------------------------------------------------
extern "C" void kernel_launch(void* const* d_in, const int* in_sizes, int n_in,
                              void* d_out, int out_size, void* d_ws, size_t ws_size,
                              hipStream_t stream) {
  const float* batch  = (const float*)d_in[0];
  const float* weight = (const float*)d_in[1];
  const float* bias   = (const float*)d_in[2];
  const int*   in_idx = (const int*)d_in[3];
  const int*   out_idx= (const int*)d_in[4];
  float*       out    = (float*)d_out;

  (void)in_sizes; (void)n_in; (void)out_size;

  const int n4 = (M_TOTAL * OUT_F) / 4;
  zero_fill_kernel<<<n4 / 256, 256, 0, stream>>>(out, n4);

  const size_t needX = (size_t)M_TOTAL * K_DIM * sizeof(unsigned short); // 64 MiB
  const size_t needW = (size_t)N_DIM   * K_DIM * sizeof(unsigned short); //  8 MiB
  dim3 grid(N_DIM / BN, M_TOTAL / BM);                                   // (16, 128)

  if (ws_size >= needX + needW) {
    unsigned short* Xb = (unsigned short*)d_ws;
    unsigned short* Wb = Xb + (size_t)M_TOTAL * K_DIM;
    convert_x_kernel<<<(M_TOTAL * (K_DIM / 8)) / 256, 256, 0, stream>>>(batch, in_idx, Xb);
    convert_w_kernel<<<(N_DIM   * (K_DIM / 8)) / 256, 256, 0, stream>>>(weight, Wb);
    gemm_bf16_wmma_kernel<<<grid, 256, 0, stream>>>(Xb, Wb, bias, out_idx, out);
  } else {
    sparse_linear_wmma_fused_kernel<<<grid, 256, 0, stream>>>(
        batch, weight, bias, in_idx, out_idx, out);
  }
}
